// AFCNet_52158082842855
// MI455X (gfx1250) — matrix operand order repaired and checked by reading, not compiled, not run on hardware
//
#include <hip/hip_runtime.h>
#include <hip/hip_bf16.h>

typedef __bf16 bf16_t;
typedef __attribute__((ext_vector_type(16))) __bf16 v16bf;
typedef __attribute__((ext_vector_type(8)))  __bf16 v8bf;
typedef __attribute__((ext_vector_type(8)))  float  v8f;
typedef __attribute__((ext_vector_type(4)))  float  v4f;

#define NSPAT 64   // 8x8 spatial positions = GEMM N dimension
#define BATCH 64

// ---------------------------------------------------------------------------
// Prologue: x (B,1024,8,8) fp32 -> xT (B, 64, 1024) bf16 (transposed so the
// WMMA B-fragment load is two lane-contiguous 16B pieces).
// ---------------------------------------------------------------------------
__global__ __launch_bounds__(256) void xpose_kernel(const float* __restrict__ x,
                                                    bf16_t* __restrict__ xT) {
  size_t i = (size_t)blockIdx.x * 256 + threadIdx.x;   // total 64*1024*64
  int n = (int)(i & 63);
  int k = (int)((i >> 6) & 1023);
  int b = (int)(i >> 16);
  xT[((size_t)b * NSPAT + n) * 1024 + k] = (bf16_t)x[i];
}

// ---------------------------------------------------------------------------
// Batched GEMM with fused bias + sigmoid + optional dropout.
//   W    : (B, cout, cin) fp32, streamed from HBM, cvt->bf16 in registers
//   XT   : (B, 64, cin)   bf16 transposed activations
//   drop : (B, cout, 64)  fp32 uniform values or nullptr
//   OT   : (B, 64, cout)  bf16 transposed output (next layer's XT)
// One wave = one 16x16 output tile; K stepped by 32 via v_wmma_f32_16x16x32_bf16.
// ---------------------------------------------------------------------------
__global__ __launch_bounds__(256) void gemm_bf16_wmma(
    const float* __restrict__ W, const float* __restrict__ bias,
    const bf16_t* __restrict__ XT, const float* __restrict__ drop,
    bf16_t* __restrict__ OT, int cout, int cin)
{
  const int lane = threadIdx.x & 31;
  const int wid  = blockIdx.x * 8 + (threadIdx.x >> 5);
  const int tiles_m  = cout >> 4;
  const int tiles_pb = tiles_m * (NSPAT / 16);     // tiles per batch
  const int b  = wid / tiles_pb;
  const int t  = wid - b * tiles_pb;
  const int mt = t >> 2;                           // 16-row tile index
  const int nt = t & 3;                            // 16-col tile index
  const int half = lane >> 4;                      // lane half selects K sub-block
  const int l15  = lane & 15;

  // A: lane l loads row (mt*16 + l15); 16-bit A layout wants K = half*8..+7 and
  //    K = 16 + half*8..+7 within each 32-wide K step -> two contiguous 32B loads.
  const float*  wrow = W  + ((size_t)b * cout + (mt * 16 + l15)) * cin;
  // B: lane l loads column (nt*16 + l15); matches A: K = half*8..+7 and
  //    K = 16 + half*8..+7 -> two 16B bf16 loads from the transposed activation tensor.
  const bf16_t* xrow = XT + ((size_t)b * NSPAT + (nt * 16 + l15)) * cin;

  v8f acc = {0.f, 0.f, 0.f, 0.f, 0.f, 0.f, 0.f, 0.f};

  for (int k0 = 0; k0 < cin; k0 += 32) {
    const float* pa0 = wrow + k0 + half * 8;
    const float* pa1 = wrow + k0 + 16 + half * 8;
    v4f a0 = *(const v4f*)(pa0);
    v4f a1 = *(const v4f*)(pa0 + 4);
    v4f a2 = *(const v4f*)(pa1);
    v4f a3 = *(const v4f*)(pa1 + 4);
    v16bf Af;
#pragma unroll
    for (int i = 0; i < 4; ++i) {
      Af[i]      = (bf16_t)a0[i];
      Af[4 + i]  = (bf16_t)a1[i];
      Af[8 + i]  = (bf16_t)a2[i];
      Af[12 + i] = (bf16_t)a3[i];
    }

    const bf16_t* pb = xrow + k0 + half * 8;
    v8bf bl0 = *(const v8bf*)(pb);
    v8bf bl1 = *(const v8bf*)(pb + 16);
    v16bf Bf;
#pragma unroll
    for (int i = 0; i < 8; ++i) { Bf[i] = bl0[i]; Bf[8 + i] = bl1[i]; }

    // D = A x B + C, fp32 accumulate
    acc = __builtin_amdgcn_wmma_f32_16x16x32_bf16(
        /*neg_a=*/false, Af, /*neg_b=*/false, Bf,
        /*c_mod=*/(short)0, acc, /*reuse_a=*/false, /*reuse_b=*/false);
  }

  // Epilogue: D layout -> lane holds column ncol, 8 contiguous rows mbase..+7.
  const int mbase = mt * 16 + half * 8;
  const int ncol  = nt * 16 + l15;
  const float* bp = bias + (size_t)b * cout + mbase;

  v8bf outv;
#pragma unroll
  for (int g = 0; g < 8; ++g) {
    float v = acc[g] + bp[g];
    v = 1.0f / (1.0f + __expf(-v));                       // sigmoid
    if (drop) {
      float dm = drop[((size_t)b * cout + (mbase + g)) * NSPAT + ncol];
      v = (dm >= 0.5f) ? v * 2.0f : 0.0f;                 // inverted dropout, p=0.5
    }
    outv[g] = (bf16_t)v;
  }
  // Single 16B store, already in the transposed layout the next layer reads.
  *(v8bf*)(OT + ((size_t)b * NSPAT + ncol) * cout + mbase) = outv;
}

// ---------------------------------------------------------------------------
// L5: cout == 1 -> plain fp32 dot product per output element (4096 outputs).
// ---------------------------------------------------------------------------
__global__ __launch_bounds__(256) void layer5_kernel(
    const float* __restrict__ w5, const float* __restrict__ b5,
    const bf16_t* __restrict__ q4T, float* __restrict__ out)
{
  int idx = blockIdx.x * 256 + threadIdx.x;     // 0..4095
  int b = idx >> 6, n = idx & 63;
  const bf16_t* q = q4T + ((size_t)b * NSPAT + n) * 64;
  const float*  w = w5 + (size_t)b * 64;
  float s = b5[b];
#pragma unroll 8
  for (int k = 0; k < 64; ++k) s += (float)q[k] * w[k];
  out[idx] = s;
}

// ---------------------------------------------------------------------------
extern "C" void kernel_launch(void* const* d_in, const int* in_sizes, int n_in,
                              void* d_out, int out_size, void* d_ws, size_t ws_size,
                              hipStream_t stream) {
  (void)in_sizes; (void)n_in; (void)out_size; (void)ws_size;
  const float* x     = (const float*)d_in[0];
  const float* w1    = (const float*)d_in[1];
  const float* b1    = (const float*)d_in[2];
  const float* w2    = (const float*)d_in[3];
  const float* b2    = (const float*)d_in[4];
  const float* w3    = (const float*)d_in[5];
  const float* b3    = (const float*)d_in[6];
  const float* w4    = (const float*)d_in[7];
  const float* b4    = (const float*)d_in[8];
  const float* w5    = (const float*)d_in[9];
  const float* b5    = (const float*)d_in[10];
  const float* drop1 = (const float*)d_in[11];
  const float* drop2 = (const float*)d_in[12];

  char* ws = (char*)d_ws;
  bf16_t* xT  = (bf16_t*)(ws);                 // 64*64*1024*2 = 8388608 B
  bf16_t* q1T = (bf16_t*)(ws + 8388608);       // 64*64*512*2  = 4194304 B
  bf16_t* q2T = (bf16_t*)(ws + 12582912);      // 64*64*256*2  = 2097152 B
  bf16_t* q3T = (bf16_t*)(ws + 14680064);      // 64*64*128*2  = 1048576 B
  bf16_t* q4T = (bf16_t*)(ws + 15728640);      // 64*64*64*2   =  524288 B

  // 1) transpose + bf16-convert input:  4,194,304 elems / 256
  xpose_kernel<<<16384, 256, 0, stream>>>(x, xT);
  // 2) L1: cout=512, cin=1024 -> 32*4*64 = 8192 waves / 8 = 1024 blocks
  gemm_bf16_wmma<<<1024, 256, 0, stream>>>(w1, b1, xT,  drop1,   q1T, 512, 1024);
  // 3) L2: cout=256, cin=512  -> 4096 waves -> 512 blocks
  gemm_bf16_wmma<<< 512, 256, 0, stream>>>(w2, b2, q1T, drop2,   q2T, 256, 512);
  // 4) L3: cout=128, cin=256  -> 2048 waves -> 256 blocks
  gemm_bf16_wmma<<< 256, 256, 0, stream>>>(w3, b3, q2T, nullptr, q3T, 128, 256);
  // 5) L4: cout=64,  cin=128  -> 1024 waves -> 128 blocks
  gemm_bf16_wmma<<< 128, 256, 0, stream>>>(w4, b4, q3T, nullptr, q4T,  64, 128);
  // 6) L5: scalar dot products into fp32 output (B,8,8)
  layer5_kernel<<<16, 256, 0, stream>>>(w5, b5, q4T, (float*)d_out);
}
